// MultiHeadAttention_7267084665156
// MI455X (gfx1250) — matrix operand order repaired and checked by reading, not compiled, not run on hardware
//
#include <hip/hip_runtime.h>
#include <hip/hip_bf16.h>

// ---------------------------------------------------------------------------
// MultiHeadAttention with relative-position bias, CDNA5 (gfx1250, wave32)
// B=4, S=2048, D=256, H=8, d_k=32.  bf16 WMMA (v_wmma_f32_16x16x32_bf16),
// flash-attention style fused softmax (no HBM score materialization),
// double-buffered K/V staging, quad-parallel softmax with wave shuffles.
// ---------------------------------------------------------------------------

typedef __attribute__((ext_vector_type(16))) __bf16          v16bf;
typedef __attribute__((ext_vector_type(16))) unsigned short  v16u16;
typedef __attribute__((ext_vector_type(8)))  float           v8f;

union U4 { uint4 q; unsigned short s[8]; };

__device__ __forceinline__ unsigned short f2bf(float f) {
    unsigned int u = __float_as_uint(f);
    unsigned int r = u + 0x7FFFu + ((u >> 16) & 1u);   // round-to-nearest-even
    return (unsigned short)(r >> 16);
}

__device__ __forceinline__ v8f wmma_bf16(v16bf a, v16bf b, v8f c) {
    // (neg_a, A, neg_b, B, c_mod, C, reuse_a, reuse_b)
    return __builtin_amdgcn_wmma_f32_16x16x32_bf16(false, a, false, b,
                                                   (short)0, c, false, false);
}

// A-matrix 16x32 bf16 fragment (ISA 7.12.2): lane L (m=L&15, hk=L>>4) holds
// k = 8*hk + {0..7} (elems 0..7) and k = 8*hk + 16 + {0..7} (elems 8..15).
__device__ __forceinline__ v16bf load_afrag(const unsigned short* base, int stride) {
    int lane = threadIdx.x & 31;
    int m = lane & 15, hk = lane >> 4;
    const unsigned short* p = base + m * stride + hk * 8;
    v16u16 u;
#pragma unroll
    for (int e = 0; e < 8; ++e) u[e] = p[e];
#pragma unroll
    for (int e = 0; e < 8; ++e) u[8 + e] = p[16 + e];
    return __builtin_bit_cast(v16bf, u);
}

// B-matrix 32x16 bf16 fragment: B[k][n] = base[n*stride + k]; lane L
// (n=L&15, kh=L>>4) holds k = 8*kh + {0..7} (elems 0..7) and k = 8*kh + 16 + {0..7} (elems 8..15).
__device__ __forceinline__ v16bf load_bfrag(const unsigned short* base, int stride) {
    int lane = threadIdx.x & 31;
    int n = lane & 15, kh = lane >> 4;
    const unsigned short* p = base + n * stride + kh * 8;
    v16u16 u;
#pragma unroll
    for (int e = 0; e < 8; ++e) u[e] = p[e];
#pragma unroll
    for (int e = 0; e < 8; ++e) u[8 + e] = p[16 + e];
    return __builtin_bit_cast(v16bf, u);
}

// ---------------------------------------------------------------------------
// GEMM: C[m][n] = (sum_k A[m][k] * W[n][k] + bias[n]) * scale
// M=8192, N=K=256.  Block = 256 thr (8 waves), 64x64 C tile,
// wave w -> subtiles (mi = w&3, nj = (w>>2)*2 + {0,1}).
// OUT_HEAD: write bf16 to [B,H,S,32]; else fp32 row-major.
// ---------------------------------------------------------------------------
template <bool A_BF16, bool OUT_HEAD>
__global__ __launch_bounds__(256) void gemm256(
    const float* __restrict__ Af, const unsigned short* __restrict__ Ah,
    const float* __restrict__ W, const float* __restrict__ bias,
    float scale, unsigned short* __restrict__ outH, float* __restrict__ outF)
{
    const int K = 256;
    __shared__ unsigned short As[64 * 40];   // 64 rows x 32 k, stride 40 (80B)
    __shared__ unsigned short Bs[64 * 40];

    const int tid   = threadIdx.x;
    const int mbase = blockIdx.x * 64;
    const int nbase = blockIdx.y * 64;
    const int wid   = tid >> 5, lane = tid & 31;
    const int mi    = wid & 3,  nj   = (wid >> 2) * 2;
    const int r     = tid >> 2, c8   = (tid & 3) * 8;

    v8f acc0 = {}; v8f acc1 = {};

    for (int kb = 0; kb < K; kb += 32) {
        if (A_BF16) {
            const unsigned short* p = Ah + (size_t)(mbase + r) * K + kb + c8;
#pragma unroll
            for (int i = 0; i < 8; ++i) As[r * 40 + c8 + i] = p[i];
        } else {
            const float* p = Af + (size_t)(mbase + r) * K + kb + c8;
#pragma unroll
            for (int i = 0; i < 8; ++i) As[r * 40 + c8 + i] = f2bf(p[i]);
        }
        const float* pw = W + (size_t)(nbase + r) * K + kb + c8;
#pragma unroll
        for (int i = 0; i < 8; ++i) Bs[r * 40 + c8 + i] = f2bf(pw[i]);
        __syncthreads();

        v16bf a  = load_afrag(As + mi * 16 * 40, 40);
        v16bf b0 = load_bfrag(Bs + nj * 16 * 40, 40);
        v16bf b1 = load_bfrag(Bs + (nj + 1) * 16 * 40, 40);
        acc0 = wmma_bf16(a, b0, acc0);
        acc1 = wmma_bf16(a, b1, acc1);
        __syncthreads();
    }

    const int rr = 8 * (lane >> 4);
    const int cl = lane & 15;
#pragma unroll
    for (int j = 0; j < 2; ++j) {
        v8f cc = j ? acc1 : acc0;
        int col = nbase + (nj + j) * 16 + cl;
        float bv = bias[col];
#pragma unroll
        for (int t = 0; t < 8; ++t) {
            int m = mbase + mi * 16 + rr + t;
            float val = (cc[t] + bv) * scale;
            if (OUT_HEAD) {
                int bidx = m >> 11, s = m & 2047;
                int h = col >> 5, d = col & 31;
                outH[(((size_t)(bidx * 8 + h) * 2048 + s) << 5) + d] = f2bf(val);
            } else {
                outF[(size_t)m * 256 + col] = val;
            }
        }
    }
}

// ---------------------------------------------------------------------------
// Fused flash attention per (b,h): 64-query block per workgroup, 64-key tiles,
// double-buffered K/V, quad-parallel online softmax with fused forward bias.
// ---------------------------------------------------------------------------
__global__ __launch_bounds__(256) void attn_kernel(
    const unsigned short* __restrict__ Q, const unsigned short* __restrict__ Kb,
    const unsigned short* __restrict__ Vb, const int* __restrict__ rel_pos,
    const float* __restrict__ emb_f, const float* __restrict__ emb_b,
    unsigned short* __restrict__ outA)
{
    const int S = 2048;
    __shared__ unsigned short Qs[64 * 40];
    __shared__ unsigned short Ks[2][64 * 40];
    __shared__ unsigned short Vt[2][32 * 72];   // V transposed: [d_k][key]
    __shared__ float          Sc[64 * 65];      // score tile f32
    __shared__ unsigned short Ps[64 * 72];      // probs bf16
    __shared__ float mrow[64], lrow[64], arow[64];
    __shared__ float efh[16], ebh[16];          // per-head embedding columns

    const int tid   = threadIdx.x;
    const int bh    = blockIdx.y;               // b*8 + h
    const int b     = bh >> 3, h = bh & 7;
    const int qbase = blockIdx.x * 64;
    const size_t headoff = (size_t)bh * S * 32;
    const size_t rpbase  = (size_t)b * S * S;
    const int wid = tid >> 5, lane = tid & 31;
    const int mi  = wid & 3,  nj   = wid >> 2;  // O subtile: rows mi*16, cols nj*16
    const int r   = tid >> 2, c8   = (tid & 3) * 8;
    const int cb  = (tid & 3) * 16;

    // stage Q tile (bf16, pre-scaled by 1/sqrt(d_k))
    {
        uint4 qv = *(const uint4*)(Q + headoff + (size_t)(qbase + r) * 32 + c8);
        *(uint4*)&Qs[r * 40 + c8] = qv;
    }
    if (tid < 64) { mrow[tid] = -3.0e38f; lrow[tid] = 0.0f; arow[tid] = 0.0f; }
    if (tid < 10) { efh[tid] = emb_f[tid * 8 + h]; ebh[tid] = emb_b[tid * 8 + h]; }

    // prefetch first K/V tile into registers
    U4 kx, vx;
    kx.q = *(const uint4*)(Kb + headoff + (size_t)r * 32 + c8);
    vx.q = *(const uint4*)(Vb + headoff + (size_t)r * 32 + c8);

    v8f oc = {};
    __syncthreads();

    // hoisted, loop-invariant Q A-fragment for this wave's score rows
    const v16bf aq = load_afrag(Qs + mi * 16 * 40, 40);

    for (int kt = 0; kt < S; kt += 64) {
        const int buf = (kt >> 6) & 1;

        // ---- commit prefetched K/V registers into LDS buffers -----------
        *(uint4*)&Ks[buf][r * 40 + c8] = kx.q;
#pragma unroll
        for (int i = 0; i < 8; ++i) Vt[buf][(c8 + i) * 72 + r] = vx.s[i];
        __syncthreads();   // also orders previous iteration's PV reads

        // ---- prefetch next K/V tile (latency hidden behind compute) -----
        if (kt + 64 < S) {
            size_t off = headoff + (size_t)(kt + 64 + r) * 32 + c8;
            kx.q = *(const uint4*)(Kb + off);
            vx.q = *(const uint4*)(Vb + off);
        }
        const int* rp1 = rel_pos + rpbase + (size_t)(qbase + r) * S + kt + cb;
        const int* rp2 = rel_pos + rpbase + (size_t)(kt + r) * S + qbase + cb;
        __builtin_prefetch(rp1, 0, 1);
        __builtin_prefetch(rp2, 0, 1);

        // ---- S = Q K^T : each wave does 2 of the 16 subtiles ------------
        {
            const int snj = (wid >> 2) * 2;
#pragma unroll
            for (int j = 0; j < 2; ++j) {
                v16bf bf = load_bfrag(&Ks[buf][(snj + j) * 16 * 40], 40);
                v8f s = {};
                s = wmma_bf16(aq, bf, s);
                int colb = (snj + j) * 16 + (lane & 15);
                int rb   = mi * 16 + 8 * (lane >> 4);
#pragma unroll
                for (int t = 0; t < 8; ++t) Sc[(rb + t) * 65 + colb] = s[t];
            }
        }
        __syncthreads();

        // ---- backward (transposed) bias: coalesced rel_pos rows ---------
        {
#pragma unroll
            for (int i = 0; i < 16; ++i)
                Sc[(cb + i) * 65 + r] += ebh[rp2[i]];
        }
        __syncthreads();

        // ---- online softmax: 4 threads/row, fused forward bias ----------
        {
            float vals[16];
            float pmax = -3.0e38f;
#pragma unroll
            for (int i = 0; i < 16; ++i) {
                float v = Sc[r * 65 + cb + i] + efh[rp1[i]];
                vals[i] = v;
                pmax = fmaxf(pmax, v);
            }
            pmax = fmaxf(pmax, __shfl_xor(pmax, 1, 32));
            pmax = fmaxf(pmax, __shfl_xor(pmax, 2, 32));
            float m0  = mrow[r];
            float rmx = fmaxf(m0, pmax);
            float psum = 0.0f;
#pragma unroll
            for (int i = 0; i < 16; ++i) {
                float p = __expf(vals[i] - rmx);
                Ps[r * 72 + cb + i] = f2bf(p);
                psum += p;
            }
            psum += __shfl_xor(psum, 1, 32);
            psum += __shfl_xor(psum, 2, 32);
            if ((tid & 3) == 0) {
                float alpha = __expf(m0 - rmx);
                mrow[r] = rmx;
                lrow[r] = lrow[r] * alpha + psum;
                arow[r] = alpha;
            }
        }
        __syncthreads();

        // ---- rescale O accumulator, then O += P @ V ---------------------
        {
            int rb = mi * 16 + 8 * (lane >> 4);
#pragma unroll
            for (int t = 0; t < 8; ++t) oc[t] *= arow[rb + t];
#pragma unroll
            for (int kc = 0; kc < 2; ++kc) {
                v16bf a  = load_afrag(Ps + mi * 16 * 72 + kc * 32, 72);
                v16bf bf = load_bfrag(&Vt[buf][nj * 16 * 72 + kc * 32], 72);
                oc = wmma_bf16(a, bf, oc);
            }
        }
        // no trailing barrier: next iteration's post-store barrier plus the
        // alternate K/V buffer provide the required ordering.
    }

    // ---- finalize: O /= l, write bf16 to [B,S,H,32] ---------------------
    {
        int rb = mi * 16 + 8 * (lane >> 4);
        int d  = nj * 16 + (lane & 15);
#pragma unroll
        for (int t = 0; t < 8; ++t) {
            float v = oc[t] / lrow[rb + t];
            int srow = qbase + rb + t;
            outA[(((size_t)b * 2048 + srow) * 8 + h) * 32 + d] = f2bf(v);
        }
    }
}

// ---------------------------------------------------------------------------
extern "C" void kernel_launch(void* const* d_in, const int* in_sizes, int n_in,
                              void* d_out, int out_size, void* d_ws, size_t ws_size,
                              hipStream_t stream) {
    (void)in_sizes; (void)n_in; (void)out_size; (void)ws_size;
    const float* query  = (const float*)d_in[0];
    const float* key    = (const float*)d_in[1];
    const float* value  = (const float*)d_in[2];
    const int*   relpos = (const int*)  d_in[3];
    const float* Wk = (const float*)d_in[4];
    const float* bk = (const float*)d_in[5];
    const float* Wv = (const float*)d_in[6];
    const float* bv = (const float*)d_in[7];
    const float* Wq = (const float*)d_in[8];
    const float* bq = (const float*)d_in[9];
    const float* Wp = (const float*)d_in[10];
    const float* bp = (const float*)d_in[11];
    const float* embF = (const float*)d_in[12];
    const float* embB = (const float*)d_in[13];

    // workspace: Q,K,V,attn-out, each [B,H,S,32] / [B,S,H,32] bf16 = 4 MB
    const size_t HEAD_ELEMS = (size_t)4 * 8 * 2048 * 32;   // 2,097,152
    unsigned short* Qbf = (unsigned short*)d_ws;
    unsigned short* Kbf = Qbf + HEAD_ELEMS;
    unsigned short* Vbf = Kbf + HEAD_ELEMS;
    unsigned short* Abf = Vbf + HEAD_ELEMS;

    const float qscale = 0.17677669529663687f;   // 1/sqrt(d_k=32)
    dim3 gg(128, 4), bb(256);

    gemm256<false, true><<<gg, bb, 0, stream>>>(query, nullptr, Wq, bq, qscale, Qbf, nullptr);
    gemm256<false, true><<<gg, bb, 0, stream>>>(key,   nullptr, Wk, bk, 1.0f,   Kbf, nullptr);
    gemm256<false, true><<<gg, bb, 0, stream>>>(value, nullptr, Wv, bv, 1.0f,   Vbf, nullptr);

    attn_kernel<<<dim3(32, 32), bb, 0, stream>>>(Qbf, Kbf, Vbf, relpos, embF, embB, Abf);

    gemm256<true, false><<<gg, bb, 0, stream>>>(nullptr, Abf, Wp, bp, 1.0f, nullptr, (float*)d_out);
}
